// VM2Model_17901423690384
// MI455X (gfx1250) — matrix operand order repaired and checked by reading, not compiled, not run on hardware
//
#include <hip/hip_runtime.h>
#include <hip/hip_bf16.h>

typedef __attribute__((ext_vector_type(16))) _Float16 v16h;
typedef __attribute__((ext_vector_type(8)))  float    v8f;
typedef __attribute__((ext_vector_type(4)))  int      v4i;

#define SEQ_LEN   256
#define VOCAB     96
#define BATCH     2048
#define DIM       (SEQ_LEN * VOCAB)        // 24576
#define NT        6                        // vocab tiles of 16
#define KC        3                        // K chunks of 32 (covers 96)
#define FRAGS_S   (NT * KC)                // 18 B-fragments per seq slice
#define HALF_S    (FRAGS_S * 32 * 16)      // 9216 halves = 18KB per slice
#define LDS_VEC   (HALF_S / 8)             // 1152 uint4 per slice
#define TBL_ELEMS (SEQ_LEN * HALF_S)       // 2359296 == 96*24576
#define MAX_SPLIT 16                       // up to 256 WGs in phase 2

#if defined(__HIP_DEVICE_COMPILE__) && \
    __has_builtin(__builtin_amdgcn_global_load_async_to_lds_b128)
#define HAVE_ASYNC 1
typedef __attribute__((address_space(1))) v4i as1_v4i;
typedef __attribute__((address_space(3))) v4i as3_v4i;
#else
#define HAVE_ASYNC 0
#endif

// ---------------------------------------------------------------------------
// Phase 1: log-weights -> f16, packed in WMMA B-fragment order:
//   t = ((s*18 + nt*3 + kc)*32 + lane)*16 + h
//   element: v = nt*16 + (lane&15), K = (h&7) + 8*(lane>=16) + 16*(h>>3), c = kc*32 + K,
//            d = s*96 + c, source = raw[v*DIM + d]
// ---------------------------------------------------------------------------
__global__ void build_logw_frags(const float* __restrict__ raw,
                                 _Float16* __restrict__ tbl)
{
    int t = blockIdx.x * 256 + threadIdx.x;
    if (t >= TBL_ELEMS) return;
    int h    = t & 15;
    int lane = (t >> 4) & 31;
    int frag = (t >> 9) % FRAGS_S;
    int s    = t / (FRAGS_S * 512);
    int nt   = frag / KC;
    int kc   = frag % KC;
    int v    = nt * 16 + (lane & 15);
    int K    = (h & 7) + ((lane >= 16) ? 8 : 0) + ((h >> 3) << 4);
    int d    = s * VOCAB + kc * 32 + K;

    float x  = raw[(size_t)v * DIM + d];
    float w  = 2.0f / (1.0f + __expf(-x));         // 2*sigmoid(x)
    float lw = __logf(fmaxf(w, 1e-8f));
    tbl[t] = (_Float16)lw;
}

// ---------------------------------------------------------------------------
// Phase 2: one-hot WMMA GEMM over a seq-split.
//   block = 8 wave32; each wave owns 16 batch rows; block owns 128 rows.
//   gridDim = (16 batch tiles, SPLIT seq splits)
// ---------------------------------------------------------------------------
union AFrag { v16h v; unsigned int u[8]; };

__device__ __forceinline__ void issue_fill(const uint4* __restrict__ src,
                                           uint4* dst, int tid)
{
#if HAVE_ASYNC
    for (int i = tid; i < LDS_VEC; i += 256)
        __builtin_amdgcn_global_load_async_to_lds_b128(
            (as1_v4i*)(src + i), (as3_v4i*)(dst + i), 0, 0);
#endif
}

__device__ __forceinline__ void wait_async()
{
#if HAVE_ASYNC
#if __has_builtin(__builtin_amdgcn_s_wait_asynccnt)
    __builtin_amdgcn_s_wait_asynccnt(0);
#else
    asm volatile("s_wait_asynccnt 0x0" ::: "memory");
#endif
#endif
}

__global__ void __launch_bounds__(256) onehot_wmma(
    const int* __restrict__ cv, const _Float16* __restrict__ tbl,
    float* __restrict__ lsum, float* __restrict__ nactBuf, int sPerSplit)
{
    __shared__ __align__(32) uint4 smem4[2 * LDS_VEC];   // 2 x 18KB slices

    const int tid    = threadIdx.x;
    const int lane   = tid & 31;
    const int wave   = tid >> 5;
    const int b0     = blockIdx.x * 128 + wave * 16;
    const int r      = lane & 15;
    const int laneHi = (lane >= 16) ? 1 : 0;
    const int split  = blockIdx.y;
    const int s0     = split * sPerSplit;

    v8f acc[NT] = {};
    float nact = 0.0f;
    const size_t rowBase = (size_t)(b0 + r) * SEQ_LEN;

#if HAVE_ASYNC
    issue_fill((const uint4*)(tbl + (size_t)s0 * HALF_S), smem4, tid);
#endif

    for (int si = 0; si < sPerSplit; ++si) {
        const int s = s0 + si;
#if HAVE_ASYNC
        uint4* cur = smem4 + (si & 1) * LDS_VEC;
        wait_async();                  // my async LDS writes landed
        __syncthreads();               // everyone's writes landed; buf ready
        if (si + 1 < sPerSplit)        // start filling the other buffer
            issue_fill((const uint4*)(tbl + (size_t)(s + 1) * HALF_S),
                       smem4 + ((si + 1) & 1) * LDS_VEC, tid);
#else
        uint4* cur = smem4;
        __syncthreads();               // protect LDS reuse
        {
            const uint4* src = (const uint4*)(tbl + (size_t)s * HALF_S);
            for (int i = tid; i < LDS_VEC; i += 256)
                cur[i] = src[i];
            if (si + 1 < sPerSplit)
                __builtin_prefetch(tbl + (size_t)(s + 1) * HALF_S + tid * 64,
                                   0, 0);
        }
        __syncthreads();
#endif
        const _Float16* sm = (const _Float16*)cur;

        const int c = cv[rowBase + s] - 1;               // -1 => masked
        nact += (c >= 0) ? 1.0f : 0.0f;

        // One-hot A fragments (16-bit A 16x32 layout):
        //   lane<16 : halves 0..7 -> K=0..7,  halves 8..15 -> K=16..23
        //   lane>=16: halves 0..7 -> K=8..15, halves 8..15 -> K=24..31
        const int kcA = c >> 5;                          // -1 if masked
        const int K   = c & 31;
        const int Kp  = K - laneHi * 8;
        int h = -1;
        if (c >= 0) {
            if (Kp >= 0 && Kp < 8)        h = Kp;
            else if (Kp >= 16 && Kp < 24) h = Kp - 8;
        }
        const unsigned int val = (h & 1) ? 0x3C000000u : 0x00003C00u;
        const int hj = h >> 1;                           // -1 if h == -1

        AFrag A[KC];
        #pragma unroll
        for (int kc = 0; kc < KC; ++kc) {
            const bool on = (kc == kcA);
            #pragma unroll
            for (int j = 0; j < 8; ++j)
                A[kc].u[j] = (on && hj == j) ? val : 0u;
        }

        // 18 WMMAs, kc-outer (A register-stationary), depth-1 pipelined B.
        v16h bcur = *(const v16h*)(sm + (size_t)lane * 16);   // frag nt=0,kc=0
        #pragma unroll
        for (int f = 0; f < FRAGS_S; ++f) {
            const int kc = f / NT, nt = f % NT;
            v16h bnext = bcur;
            if (f + 1 < FRAGS_S) {
                const int kc2 = (f + 1) / NT, nt2 = (f + 1) % NT;
                bnext = *(const v16h*)(sm + ((nt2 * KC + kc2) * 32 + lane) * 16);
            }
            acc[nt] = __builtin_amdgcn_wmma_f32_16x16x32_f16(
                false, A[kc].v, false, bcur, (short)0, acc[nt], false, false);
            bcur = bnext;
        }
    }

    // Partial results: disjoint per (split, batch-tile) -> no atomics needed.
    if (lane < 16)
        nactBuf[(size_t)split * BATCH + b0 + r] = nact;

    // C/D layout: lane holds N = lane&15; VGPR j holds M = j + 8*(lane>=16)
    #pragma unroll
    for (int j = 0; j < 8; ++j) {
        const int row = j + 8 * laneHi;
        #pragma unroll
        for (int nt = 0; nt < NT; ++nt) {
            lsum[((size_t)split * BATCH + b0 + row) * VOCAB + nt * 16 + (lane & 15)]
                = acc[nt][j];
        }
    }
}

// ---------------------------------------------------------------------------
// Phase 3: reduce seq-splits, geometric-mean finalize.
// ---------------------------------------------------------------------------
__global__ void finalize_logits(const float* __restrict__ lsum,
                                const float* __restrict__ nactBuf,
                                float* __restrict__ out, int nsplit)
{
    int i = blockIdx.x * 256 + threadIdx.x;
    if (i >= BATCH * VOCAB) return;
    int b = i / VOCAB;
    float s = 0.0f, n = 0.0f;
    for (int p = 0; p < nsplit; ++p) {
        s += lsum[(size_t)p * BATCH * VOCAB + i];
        n += nactBuf[(size_t)p * BATCH + b];
    }
    out[i] = __expf(s / fmaxf(n, 1.0f));
}

// ---------------------------------------------------------------------------
extern "C" void kernel_launch(void* const* d_in, const int* in_sizes, int n_in,
                              void* d_out, int out_size, void* d_ws, size_t ws_size,
                              hipStream_t stream)
{
    const int*   cv  = (const int*)d_in[0];     // (2048, 256) int32
    const float* raw = (const float*)d_in[1];   // (96, 24576) f32
    float*       out = (float*)d_out;           // (2048, 96)  f32

    _Float16* tbl = (_Float16*)d_ws;
    const size_t tblBytes = (size_t)TBL_ELEMS * sizeof(_Float16); // ~4.7MB

    // Occupancy: 16*SPLIT workgroups; MI455X has O(200) WGPs, so prefer 16.
    int SPLIT = MAX_SPLIT;
    while (SPLIT > 1 &&
           tblBytes + (size_t)SPLIT * ((size_t)BATCH * VOCAB * 4 + BATCH * 4)
               > ws_size)
        SPLIT >>= 1;

    float* lsum = (float*)((char*)d_ws + tblBytes);
    float* nact = lsum + (size_t)SPLIT * BATCH * VOCAB;

    // Phase 1: build f16 log-weight table in WMMA B-fragment layout.
    build_logw_frags<<<(TBL_ELEMS + 255) / 256, 256, 0, stream>>>(raw, tbl);

    // Phase 2: one-hot WMMA GEMM partials.
    dim3 grid2(BATCH / 128, SPLIT);
    onehot_wmma<<<grid2, 256, 0, stream>>>(cv, tbl, lsum, nact,
                                           SEQ_LEN / SPLIT);

    // Phase 3: reduce + exp.
    finalize_logits<<<(BATCH * VOCAB + 255) / 256, 256, 0, stream>>>(
        lsum, nact, out, SPLIT);
}
